// LuongAttention_16707422781949
// MI455X (gfx1250) — matrix operand order repaired and checked by reading, not compiled, not run on hardware
//
#include <hip/hip_runtime.h>
#include <hip/hip_bf16.h>

#define TQ_ 2048
#define TK_ 2048
#define DD_ 1024
#define NB_ 16

typedef __attribute__((ext_vector_type(16))) __bf16          v16bf;
typedef __attribute__((ext_vector_type(8)))  float           v8f;
typedef __attribute__((ext_vector_type(16))) unsigned short  v16us;
typedef __attribute__((ext_vector_type(8)))  unsigned short  v8us;
typedef __attribute__((ext_vector_type(4)))  unsigned short  v4us;

union BfFrag { v16us u; v16bf b; };

__device__ __forceinline__ unsigned short f2bf(float f) {
  unsigned int x = __float_as_uint(f);
  x += 0x7FFFu + ((x >> 16) & 1u);          // round-to-nearest-even
  return (unsigned short)(x >> 16);
}

// ---- A fragment from bf16 row-major A[m][k]; tile (m0..m0+15, k0..k0+31).
// ISA 16-bit A 16x32 layout: lane m = lane&15; koff = (lane>=16)?8:0;
// elems 0..7 -> K = k0+koff+i ; elems 8..15 -> K = k0+16+koff+i.
__device__ __forceinline__ v16bf load_a_bf16(const unsigned short* __restrict__ src,
                                             int ld, int m0, int k0, int lane) {
  int row  = m0 + (lane & 15);
  int koff = (lane >> 4) << 3;
  const v8us* p0 = reinterpret_cast<const v8us*>(src + (size_t)row * ld + k0 + koff);
  const v8us* p1 = reinterpret_cast<const v8us*>(src + (size_t)row * ld + k0 + 16 + koff);
  v8us lo = p0[0], hi = p1[0];
  BfFrag f;
#pragma unroll
  for (int i = 0; i < 8; ++i) { f.u[i] = lo[i]; f.u[8 + i] = hi[i]; }
  return f.b;
}

// ---- B fragment: bf16 source stored [N][K] row-major; tile (n0..n0+15, k0..k0+31).
// ISA 16-bit B 32x16 layout: lane n = lane&15; koff = (lane>=16)?8:0;
// elems 0..7 -> K = k0+koff+i ; elems 8..15 -> K = k0+16+koff+i.
__device__ __forceinline__ v16bf load_b_bf16(const unsigned short* __restrict__ src,
                                             int ld, int n0, int k0, int lane) {
  int col   = n0 + (lane & 15);
  int kbase = k0 + ((lane >> 4) << 3);
  const v8us* p = reinterpret_cast<const v8us*>(src + (size_t)col * ld + kbase);
  v8us lo = p[0], hi = p[2];
  BfFrag f;
#pragma unroll
  for (int i = 0; i < 8; ++i) { f.u[i] = lo[i]; f.u[8 + i] = hi[i]; }
  return f.b;
}

// ---- elementwise fp32 -> bf16 (n multiple of 4)
__global__ __launch_bounds__(256) void convert_f32_to_bf16(
    const float* __restrict__ in, unsigned short* __restrict__ out, size_t n) {
  size_t i = ((size_t)blockIdx.x * 256 + threadIdx.x) * 4;
  if (i >= n) return;
  float4 v = *reinterpret_cast<const float4*>(in + i);
  v4us o;
  o[0] = f2bf(v.x); o[1] = f2bf(v.y); o[2] = f2bf(v.z); o[3] = f2bf(v.w);
  *reinterpret_cast<v4us*>(out + i) = o;
}

// ---- fp32 -> bf16 transpose: out[c][r] = bf16(in[r][c]); rows,cols multiples of 32.
__global__ __launch_bounds__(256) void transpose_f32_to_bf16(
    const float* __restrict__ in, unsigned short* __restrict__ out,
    int rows, int cols, size_t in_bstride, size_t out_bstride) {
  __shared__ float tile[32][33];
  const float* I = in + (size_t)blockIdx.z * in_bstride;
  unsigned short* O = out + (size_t)blockIdx.z * out_bstride;
  int r0 = blockIdx.y * 32, c0 = blockIdx.x * 32;
  int tx = threadIdx.x & 31, ty = threadIdx.x >> 5;
#pragma unroll
  for (int i = 0; i < 32; i += 8)
    tile[ty + i][tx] = I[(size_t)(r0 + ty + i) * cols + (c0 + tx)];
  __syncthreads();
#pragma unroll
  for (int i = 0; i < 32; i += 8)
    O[(size_t)(c0 + ty + i) * rows + (r0 + tx)] = f2bf(tile[tx][ty + i]);
}

// ---- Generic 32x64-per-wave bf16 GEMM body: C(32x64 tile) += A(32xK) * B(K x 64).
// A bf16 [M][K] row-major, B bf16 [N][K] row-major ("N-major"), both ld = K.
// Returns accumulators; caller handles epilogue.
struct Acc32x64 { v8f a[2][4]; };

__device__ __forceinline__ Acc32x64 gemm_32x64(
    const unsigned short* __restrict__ A, const unsigned short* __restrict__ Bm,
    int K, int m0, int n0, int lane) {
  Acc32x64 acc;
#pragma unroll
  for (int h = 0; h < 2; ++h)
#pragma unroll
    for (int t = 0; t < 4; ++t) acc.a[h][t] = (v8f){};
  for (int k0 = 0; k0 < K; k0 += 32) {
    v16bf a0 = load_a_bf16(A, K, m0, k0, lane);
    v16bf a1 = load_a_bf16(A, K, m0 + 16, k0, lane);
#pragma unroll
    for (int t = 0; t < 4; ++t) {
      v16bf bb = load_b_bf16(Bm, K, n0 + 16 * t, k0, lane);
      acc.a[0][t] = __builtin_amdgcn_wmma_f32_16x16x32_bf16(
          false, a0, false, bb, (short)0, acc.a[0][t], false, false);
      acc.a[1][t] = __builtin_amdgcn_wmma_f32_16x16x32_bf16(
          false, a1, false, bb, (short)0, acc.a[1][t], false, false);
    }
  }
  return acc;
}

// ---- Kernel 1: keys = enc @ W + b, output bf16 [NB*TK][D].
__global__ __launch_bounds__(128) void keys_gemm(
    const unsigned short* __restrict__ encb, const unsigned short* __restrict__ wt,
    const float* __restrict__ bias, unsigned short* __restrict__ keys) {
  int lane = threadIdx.x & 31;
  int wave = threadIdx.x >> 5;
  int m0 = blockIdx.y * 32;
  int n0 = blockIdx.x * 256 + wave * 64;
  Acc32x64 acc = gemm_32x64(encb, wt, DD_, m0, n0, lane);
  int n  = lane & 15;
  int mb = ((lane >> 4) << 3);
#pragma unroll
  for (int t = 0; t < 4; ++t) {
    float bv = bias[n0 + 16 * t + n];
#pragma unroll
    for (int h = 0; h < 2; ++h)
#pragma unroll
      for (int r = 0; r < 8; ++r)
        keys[(size_t)(m0 + 16 * h + mb + r) * DD_ + (n0 + 16 * t + n)] =
            f2bf(acc.a[h][t][r] + bv);
  }
}

// ---- Kernel 2: raw scores = dec @ keys^T, fp32 out [TQ][TK] per batch.
__global__ __launch_bounds__(128) void score_gemm(
    const unsigned short* __restrict__ decb, const unsigned short* __restrict__ keys,
    float* __restrict__ score) {
  int b = blockIdx.z;
  const unsigned short* A = decb + (size_t)b * TQ_ * DD_;
  const unsigned short* K = keys + (size_t)b * TK_ * DD_;
  float* O = score + (size_t)b * TQ_ * TK_;
  int lane = threadIdx.x & 31;
  int wave = threadIdx.x >> 5;
  int m0 = blockIdx.y * 32;
  int n0 = blockIdx.x * 256 + wave * 64;
  Acc32x64 acc = gemm_32x64(A, K, DD_, m0, n0, lane);
  int n  = lane & 15;
  int mb = ((lane >> 4) << 3);
#pragma unroll
  for (int t = 0; t < 4; ++t)
#pragma unroll
    for (int h = 0; h < 2; ++h)
#pragma unroll
      for (int r = 0; r < 8; ++r)
        O[(size_t)(m0 + 16 * h + mb + r) * TK_ + (n0 + 16 * t + n)] = acc.a[h][t][r];
}

// ---- Kernel 3: row softmax over TK. One wave32 per row; 64 vals/lane.
// Writes normalized fp32 in place (d_out alignment) AND bf16 copy for context GEMM.
__global__ __launch_bounds__(256) void softmax_rows(
    float* __restrict__ align, unsigned short* __restrict__ pb) {
  int lane = threadIdx.x & 31;
  int wave = threadIdx.x >> 5;
  size_t row = (size_t)blockIdx.x * 8 + wave;
  float* p = align + row * TK_;
  unsigned short* q = pb + row * TK_;
  float v[64];
  float mx = -3.402823466e+38f;
#pragma unroll
  for (int i = 0; i < 64; ++i) { v[i] = p[i * 32 + lane]; mx = fmaxf(mx, v[i]); }
#pragma unroll
  for (int off = 16; off > 0; off >>= 1) mx = fmaxf(mx, __shfl_xor(mx, off, 32));
  float s = 0.f;
#pragma unroll
  for (int i = 0; i < 64; ++i) { v[i] = __expf(v[i] - mx); s += v[i]; }
#pragma unroll
  for (int off = 16; off > 0; off >>= 1) s += __shfl_xor(s, off, 32);
  float inv = 1.0f / s;
#pragma unroll
  for (int i = 0; i < 64; ++i) {
    float r = v[i] * inv;
    p[i * 32 + lane] = r;
    q[i * 32 + lane] = f2bf(r);
  }
}

// ---- Kernel 4: context = P @ enc, fp32 out [TQ][D] per batch.
__global__ __launch_bounds__(128) void context_gemm(
    const unsigned short* __restrict__ pb, const unsigned short* __restrict__ enct,
    float* __restrict__ ctx) {
  int b = blockIdx.z;
  const unsigned short* A = pb + (size_t)b * TQ_ * TK_;
  const unsigned short* E = enct + (size_t)b * DD_ * TK_;
  float* O = ctx + (size_t)b * TQ_ * DD_;
  int lane = threadIdx.x & 31;
  int wave = threadIdx.x >> 5;
  int m0 = blockIdx.y * 32;
  int n0 = blockIdx.x * 256 + wave * 64;
  Acc32x64 acc = gemm_32x64(A, E, TK_, m0, n0, lane);
  int n  = lane & 15;
  int mb = ((lane >> 4) << 3);
#pragma unroll
  for (int t = 0; t < 4; ++t)
#pragma unroll
    for (int h = 0; h < 2; ++h)
#pragma unroll
      for (int r = 0; r < 8; ++r)
        O[(size_t)(m0 + 16 * h + mb + r) * DD_ + (n0 + 16 * t + n)] = acc.a[h][t][r];
}

extern "C" void kernel_launch(void* const* d_in, const int* in_sizes, int n_in,
                              void* d_out, int out_size, void* d_ws, size_t ws_size,
                              hipStream_t stream) {
  const float* dec  = (const float*)d_in[0];   // [NB, TQ, D]
  const float* enc  = (const float*)d_in[1];   // [NB, TK, D]
  const float* wa_w = (const float*)d_in[2];   // [D, D]
  const float* wa_b = (const float*)d_in[3];   // [D]

  float* context = (float*)d_out;                                  // [NB, TQ, D]
  float* align   = (float*)d_out + (size_t)NB_ * TQ_ * DD_;        // [NB, TQ, TK]

  unsigned short* ws    = (unsigned short*)d_ws;
  unsigned short* wt    = ws;                                      // [D][D]        2 MiB
  unsigned short* keysb = wt + (size_t)DD_ * DD_;                  // [NB*TK][D]   64 MiB
  unsigned short* enct  = keysb + (size_t)NB_ * TK_ * DD_;         // [NB][D][TK]  64 MiB
  unsigned short* decb  = enct + (size_t)NB_ * DD_ * TK_;          // [NB*TQ][D]   64 MiB
  unsigned short* encb  = decb + (size_t)NB_ * TQ_ * DD_;          // [NB*TK][D]   64 MiB
  unsigned short* pb    = encb + (size_t)NB_ * TK_ * DD_;          // [NB*TQ][TK] 128 MiB

  // W^T (bf16) — B-operand for keys GEMM, stays L2-resident (2 MiB).
  transpose_f32_to_bf16<<<dim3(DD_ / 32, DD_ / 32, 1), 256, 0, stream>>>(
      wa_w, wt, DD_, DD_, 0, 0);
  // enc^T (bf16) per batch — B-operand for context GEMM.
  transpose_f32_to_bf16<<<dim3(DD_ / 32, TK_ / 32, NB_), 256, 0, stream>>>(
      enc, enct, TK_, DD_, (size_t)TK_ * DD_, (size_t)DD_ * TK_);
  // dec/enc (bf16) — A-operands for score/keys GEMMs (read 4-8x there; convert once).
  {
    size_t n = (size_t)NB_ * TQ_ * DD_;
    convert_f32_to_bf16<<<dim3((unsigned)(n / 4 / 256), 1, 1), 256, 0, stream>>>(
        dec, decb, n);
    size_t m = (size_t)NB_ * TK_ * DD_;
    convert_f32_to_bf16<<<dim3((unsigned)(m / 4 / 256), 1, 1), 256, 0, stream>>>(
        enc, encb, m);
  }

  // keys = enc @ W + b  (bf16 out)
  keys_gemm<<<dim3(DD_ / 256, (NB_ * TK_) / 32, 1), 128, 0, stream>>>(
      encb, wt, wa_b, keysb);

  // raw scores into alignment slice of d_out
  score_gemm<<<dim3(TK_ / 256, TQ_ / 32, NB_), 128, 0, stream>>>(decb, keysb, align);

  // softmax over TK: fp32 in place + bf16 P for context GEMM
  softmax_rows<<<dim3((NB_ * TQ_) / 8, 1, 1), 256, 0, stream>>>(align, pb);

  // context = P @ enc
  context_gemm<<<dim3(DD_ / 256, TQ_ / 32, NB_), 128, 0, stream>>>(pb, enct, context);
}
